// GCN_2954937499939
// MI455X (gfx1250) — matrix operand order repaired and checked by reading, not compiled, not run on hardware
//
#include <hip/hip_runtime.h>
#include <hip/hip_bf16.h>
#include <math.h>

// Problem constants (from reference)
#define NN     8192
#define NFEAT  512
#define NHID   128
#define NCLASS 16

typedef __attribute__((ext_vector_type(16))) __bf16 v16bf;
typedef __attribute__((ext_vector_type(8)))  __bf16 v8bf;
typedef __attribute__((ext_vector_type(8)))  float  v8f;
typedef __attribute__((ext_vector_type(4)))  float  v4f;

static __device__ __forceinline__ v16bf bf_cat(v8bf lo, v8bf hi) {
  return __builtin_shufflevector(lo, hi, 0,1,2,3,4,5,6,7,8,9,10,11,12,13,14,15);
}

static __device__ __forceinline__ v16bf bf_cvt16(v4f f0, v4f f1, v4f f2, v4f f3) {
  v16bf r;
  r[0]=(__bf16)f0.x;  r[1]=(__bf16)f0.y;  r[2]=(__bf16)f0.z;  r[3]=(__bf16)f0.w;
  r[4]=(__bf16)f1.x;  r[5]=(__bf16)f1.y;  r[6]=(__bf16)f1.z;  r[7]=(__bf16)f1.w;
  r[8]=(__bf16)f2.x;  r[9]=(__bf16)f2.y;  r[10]=(__bf16)f2.z; r[11]=(__bf16)f2.w;
  r[12]=(__bf16)f3.x; r[13]=(__bf16)f3.y; r[14]=(__bf16)f3.z; r[15]=(__bf16)f3.w;
  return r;
}

static __device__ __forceinline__ v8f wmma_bf16(v16bf a, v16bf b, v8f c) {
  // D = A(16x32 bf16) * B(32x16 bf16) + C(16x16 f32)
  return __builtin_amdgcn_wmma_f32_16x16x32_bf16(false, a, false, b, (short)0, c,
                                                 false, false);
}

// ---------------------------------------------------------------------------
// K0a: deg_i = 1 + sum_j adj[i,j]; dinv_i = rsqrt(max(deg,1e-12)).
//      Fused: also writes Ah = bf16(adj) (exact: entries are 0/1).
// ---------------------------------------------------------------------------
__global__ __launch_bounds__(256) void k_deg_cvt(const float* __restrict__ adj,
                                                 float* __restrict__ dinv,
                                                 __bf16* __restrict__ Ah) {
  __shared__ float red[256];
  int row = blockIdx.x;
  const v4f* p = (const v4f*)(adj + (size_t)row * NN);
  v8bf*      q = (v8bf*)(Ah + (size_t)row * NN);
  float s = 0.f;
  for (int t = threadIdx.x; t < NN / 8; t += 256) {
    v4f a = p[2 * t], b = p[2 * t + 1];
    s += a.x + a.y + a.z + a.w + b.x + b.y + b.z + b.w;
    v8bf o;
    o[0]=(__bf16)a.x; o[1]=(__bf16)a.y; o[2]=(__bf16)a.z; o[3]=(__bf16)a.w;
    o[4]=(__bf16)b.x; o[5]=(__bf16)b.y; o[6]=(__bf16)b.z; o[7]=(__bf16)b.w;
    q[t] = o;
  }
  red[threadIdx.x] = s;
  __syncthreads();
  for (int off = 128; off > 0; off >>= 1) {
    if (threadIdx.x < off) red[threadIdx.x] += red[threadIdx.x + off];
    __syncthreads();
  }
  if (threadIdx.x == 0)
    dinv[row] = rsqrtf(fmaxf(red[0] + 1.0f, 1e-12f));
}

// K0b: fallback degree-only pass (small-workspace path)
__global__ __launch_bounds__(256) void k_deg(const float* __restrict__ adj,
                                             float* __restrict__ dinv) {
  __shared__ float red[256];
  int row = blockIdx.x;
  const v4f* p = (const v4f*)(adj + (size_t)row * NN);
  float s = 0.f;
  for (int t = threadIdx.x; t < NN / 4; t += 256) {
    v4f v = p[t];
    s += v.x + v.y + v.z + v.w;
  }
  red[threadIdx.x] = s;
  __syncthreads();
  for (int off = 128; off > 0; off >>= 1) {
    if (threadIdx.x < off) red[threadIdx.x] += red[threadIdx.x + off];
    __syncthreads();
  }
  if (threadIdx.x == 0)
    dinv[row] = rsqrtf(fmaxf(red[0] + 1.0f, 1e-12f));
}

// ---------------------------------------------------------------------------
// K1: x (f32) -> Xh (bf16), 8 elements per thread
// ---------------------------------------------------------------------------
__global__ __launch_bounds__(256) void k_cvt_x(const float* __restrict__ x,
                                               __bf16* __restrict__ xh) {
  int idx = blockIdx.x * 256 + threadIdx.x;   // one v8 per thread
  const v4f* p = (const v4f*)x;
  v4f a = p[2 * idx], b = p[2 * idx + 1];
  v8bf o;
  o[0]=(__bf16)a.x; o[1]=(__bf16)a.y; o[2]=(__bf16)a.z; o[3]=(__bf16)a.w;
  o[4]=(__bf16)b.x; o[5]=(__bf16)b.y; o[6]=(__bf16)b.z; o[7]=(__bf16)b.w;
  ((v8bf*)xh)[idx] = o;
}

// ---------------------------------------------------------------------------
// K2/K3: transpose + convert weights.  Wt[n,k] = W[k,n]
// ---------------------------------------------------------------------------
__global__ __launch_bounds__(256) void k_tr_w1(const float* __restrict__ w1,
                                               __bf16* __restrict__ w1t) {
  int idx = blockIdx.x * 256 + threadIdx.x;   // idx = n*NFEAT + k
  int n = idx / NFEAT, k = idx % NFEAT;
  w1t[idx] = (__bf16)w1[(size_t)k * NHID + n];
}

__global__ __launch_bounds__(256) void k_tr_w2(const float* __restrict__ w2,
                                               __bf16* __restrict__ w2t) {
  int idx = blockIdx.x * 256 + threadIdx.x;   // idx = n*NHID + k
  int n = idx / NHID, k = idx % NHID;
  w2t[idx] = (__bf16)w2[(size_t)k * NCLASS + n];
}

// ---------------------------------------------------------------------------
// K4: B1t[n, j] = dinv_j * (X @ W1)[j, n]   (bf16, transposed for B-fragments)
// ---------------------------------------------------------------------------
__global__ __launch_bounds__(256) void k_gemm_xw1(const __bf16* __restrict__ Xh,
                                                  const __bf16* __restrict__ W1t,
                                                  const float* __restrict__ dinv,
                                                  __bf16* __restrict__ B1t) {
  int lane = threadIdx.x & 31, wave = threadIdx.x >> 5;
  int mbase = blockIdx.x * 16;
  int n     = wave * 16 + (lane & 15);
  int m     = mbase + (lane & 15);
  int aoff  = (lane < 16) ? 0 : 8;    // K sub-group for A fragment
  int boff  = (lane < 16) ? 0 : 8;    // K sub-group for B fragment

  const __bf16* arow = Xh  + (size_t)m * NFEAT;
  const __bf16* brow = W1t + (size_t)n * NFEAT;
  v8f acc = {};
#pragma unroll 4
  for (int kb = 0; kb < NFEAT; kb += 32) {
    v8bf a0 = *(const v8bf*)(arow + kb + aoff);
    v8bf a1 = *(const v8bf*)(arow + kb + aoff + 16);
    v8bf b0 = *(const v8bf*)(brow + kb + boff);
    v8bf b1 = *(const v8bf*)(brow + kb + boff + 16);
    acc = wmma_bf16(bf_cat(a0, a1), bf_cat(b0, b1), acc);
  }
  int m0 = mbase + aoff;              // this lane's 8 output rows: m0..m0+7
  v8bf out;
#pragma unroll
  for (int r = 0; r < 8; r++) out[r] = (__bf16)(acc[r] * dinv[m0 + r]);
  *(v8bf*)(B1t + (size_t)n * NN + m0) = out;
}

// ---------------------------------------------------------------------------
// K5a: aggregation 1 from materialized bf16 adjacency (fast path)
// H[i,n] = relu(dinv_i * (sum_j A[i,j]*B1t[n,j] + B1t[n,i]) + b1[n])
// ---------------------------------------------------------------------------
__global__ __launch_bounds__(256) void k_agg1_bf(const __bf16* __restrict__ Ah,
                                                 const __bf16* __restrict__ B1t,
                                                 const float* __restrict__ dinv,
                                                 const float* __restrict__ b1,
                                                 __bf16* __restrict__ Hh) {
  int lane = threadIdx.x & 31, wave = threadIdx.x >> 5;
  int mbase = blockIdx.x * 16;
  int mrow  = mbase + (lane & 15);
  int n     = wave * 16 + (lane & 15);
  int aoff  = (lane < 16) ? 0 : 8;
  int boff  = (lane < 16) ? 0 : 8;

  const __bf16* arow = Ah  + (size_t)mrow * NN;
  const __bf16* brow = B1t + (size_t)n * NN;
  v8f acc = {};
#pragma unroll 4
  for (int kb = 0; kb < NN; kb += 32) {
    if (kb + 1024 < NN) __builtin_prefetch(arow + kb + 1024, 0, 0);
    v8bf a0 = *(const v8bf*)(arow + kb + aoff);
    v8bf a1 = *(const v8bf*)(arow + kb + aoff + 16);
    v8bf b0 = *(const v8bf*)(brow + kb + boff);
    v8bf b1v = *(const v8bf*)(brow + kb + boff + 16);
    acc = wmma_bf16(bf_cat(a0, a1), bf_cat(b0, b1v), acc);
  }
  int m0 = mbase + aoff;
#pragma unroll
  for (int r = 0; r < 8; r++) {
    int m   = m0 + r;
    float v = acc[r] + (float)B1t[(size_t)n * NN + m];   // +I self term
    float h = fmaxf(dinv[m] * v + b1[n], 0.0f);          // bias + relu
    Hh[(size_t)m * NHID + n] = (__bf16)h;
  }
}

// K5b: aggregation 1 with on-the-fly fp32->bf16 conversion (fallback)
__global__ __launch_bounds__(256) void k_agg1(const float* __restrict__ adj,
                                              const __bf16* __restrict__ B1t,
                                              const float* __restrict__ dinv,
                                              const float* __restrict__ b1,
                                              __bf16* __restrict__ Hh) {
  int lane = threadIdx.x & 31, wave = threadIdx.x >> 5;
  int mbase = blockIdx.x * 16;
  int mrow  = mbase + (lane & 15);
  int n     = wave * 16 + (lane & 15);
  int aoff  = (lane < 16) ? 0 : 8;
  int boff  = (lane < 16) ? 0 : 8;

  const float*  arow = adj + (size_t)mrow * NN;
  const __bf16* brow = B1t + (size_t)n * NN;
  v8f acc = {};
#pragma unroll 2
  for (int kb = 0; kb < NN; kb += 32) {
    if (kb + 512 < NN) __builtin_prefetch(arow + kb + 512, 0, 0);
    v4f f0 = *(const v4f*)(arow + kb + aoff);
    v4f f1 = *(const v4f*)(arow + kb + aoff + 4);
    v4f f2 = *(const v4f*)(arow + kb + aoff + 16);
    v4f f3 = *(const v4f*)(arow + kb + aoff + 20);
    v8bf b0 = *(const v8bf*)(brow + kb + boff);
    v8bf b1v = *(const v8bf*)(brow + kb + boff + 16);
    acc = wmma_bf16(bf_cvt16(f0, f1, f2, f3), bf_cat(b0, b1v), acc);
  }
  int m0 = mbase + aoff;
#pragma unroll
  for (int r = 0; r < 8; r++) {
    int m   = m0 + r;
    float v = acc[r] + (float)B1t[(size_t)n * NN + m];   // +I self term
    float h = fmaxf(dinv[m] * v + b1[n], 0.0f);          // bias + relu
    Hh[(size_t)m * NHID + n] = (__bf16)h;
  }
}

// ---------------------------------------------------------------------------
// K6: B2t[n, j] = dinv_j * (H @ W2)[j, n]   (bf16 [16, 8192])
// ---------------------------------------------------------------------------
__global__ __launch_bounds__(256) void k_gemm_hw2(const __bf16* __restrict__ Hh,
                                                  const __bf16* __restrict__ W2t,
                                                  const float* __restrict__ dinv,
                                                  __bf16* __restrict__ B2t) {
  int lane = threadIdx.x & 31, wave = threadIdx.x >> 5;
  int mbase = (blockIdx.x * 8 + wave) * 16;
  int n     = lane & 15;
  int m     = mbase + (lane & 15);
  int aoff  = (lane < 16) ? 0 : 8;
  int boff  = (lane < 16) ? 0 : 8;

  const __bf16* arow = Hh  + (size_t)m * NHID;
  const __bf16* brow = W2t + (size_t)n * NHID;
  v8f acc = {};
#pragma unroll
  for (int kb = 0; kb < NHID; kb += 32) {
    v8bf a0 = *(const v8bf*)(arow + kb + aoff);
    v8bf a1 = *(const v8bf*)(arow + kb + aoff + 16);
    v8bf b0 = *(const v8bf*)(brow + kb + boff);
    v8bf b1 = *(const v8bf*)(brow + kb + boff + 16);
    acc = wmma_bf16(bf_cat(a0, a1), bf_cat(b0, b1), acc);
  }
  int m0 = mbase + aoff;
  v8bf out;
#pragma unroll
  for (int r = 0; r < 8; r++) out[r] = (__bf16)(acc[r] * dinv[m0 + r]);
  *(v8bf*)(B2t + (size_t)n * NN + m0) = out;
}

// ---------------------------------------------------------------------------
// Shared epilogue: bias + dinv_i scale + self term + width-16 log-softmax
// ---------------------------------------------------------------------------
static __device__ __forceinline__ void agg2_epilogue(v8f acc, int m0, int n,
                                                     const __bf16* B2t,
                                                     const float* dinv,
                                                     const float* b2,
                                                     float* out) {
#pragma unroll
  for (int r = 0; r < 8; r++) {
    int m   = m0 + r;                                     // same across half-wave
    float v = acc[r] + (float)B2t[(size_t)n * NN + m];    // +I self term
    v = dinv[m] * v + b2[n];
    float mx = v;
    for (int d = 8; d > 0; d >>= 1) mx = fmaxf(mx, __shfl_xor(mx, d, 16));
    float e = expf(v - mx);
    float s = e;
    for (int d = 8; d > 0; d >>= 1) s += __shfl_xor(s, d, 16);
    out[(size_t)m * NCLASS + n] = (v - mx) - logf(s);
  }
}

// K7a: aggregation 2 from materialized bf16 adjacency (fast path)
__global__ __launch_bounds__(256) void k_agg2_bf(const __bf16* __restrict__ Ah,
                                                 const __bf16* __restrict__ B2t,
                                                 const float* __restrict__ dinv,
                                                 const float* __restrict__ b2,
                                                 float* __restrict__ out) {
  int lane = threadIdx.x & 31, wave = threadIdx.x >> 5;
  int mbase = (blockIdx.x * 8 + wave) * 16;
  int mrow  = mbase + (lane & 15);
  int n     = lane & 15;
  int aoff  = (lane < 16) ? 0 : 8;
  int boff  = (lane < 16) ? 0 : 8;

  const __bf16* arow = Ah  + (size_t)mrow * NN;
  const __bf16* brow = B2t + (size_t)n * NN;
  v8f acc = {};
#pragma unroll 4
  for (int kb = 0; kb < NN; kb += 32) {
    if (kb + 1024 < NN) __builtin_prefetch(arow + kb + 1024, 0, 0);
    v8bf a0 = *(const v8bf*)(arow + kb + aoff);
    v8bf a1 = *(const v8bf*)(arow + kb + aoff + 16);
    v8bf b0 = *(const v8bf*)(brow + kb + boff);
    v8bf b1 = *(const v8bf*)(brow + kb + boff + 16);
    acc = wmma_bf16(bf_cat(a0, a1), bf_cat(b0, b1), acc);
  }
  agg2_epilogue(acc, mbase + aoff, n, B2t, dinv, b2, out);
}

// K7b: aggregation 2 with on-the-fly conversion (fallback)
__global__ __launch_bounds__(256) void k_agg2(const float* __restrict__ adj,
                                              const __bf16* __restrict__ B2t,
                                              const float* __restrict__ dinv,
                                              const float* __restrict__ b2,
                                              float* __restrict__ out) {
  int lane = threadIdx.x & 31, wave = threadIdx.x >> 5;
  int mbase = (blockIdx.x * 8 + wave) * 16;
  int mrow  = mbase + (lane & 15);
  int n     = lane & 15;
  int aoff  = (lane < 16) ? 0 : 8;
  int boff  = (lane < 16) ? 0 : 8;

  const float*  arow = adj + (size_t)mrow * NN;
  const __bf16* brow = B2t + (size_t)n * NN;
  v8f acc = {};
#pragma unroll 2
  for (int kb = 0; kb < NN; kb += 32) {
    if (kb + 512 < NN) __builtin_prefetch(arow + kb + 512, 0, 0);
    v4f f0 = *(const v4f*)(arow + kb + aoff);
    v4f f1 = *(const v4f*)(arow + kb + aoff + 4);
    v4f f2 = *(const v4f*)(arow + kb + aoff + 16);
    v4f f3 = *(const v4f*)(arow + kb + aoff + 20);
    v8bf b0 = *(const v8bf*)(brow + kb + boff);
    v8bf b1 = *(const v8bf*)(brow + kb + boff + 16);
    acc = wmma_bf16(bf_cvt16(f0, f1, f2, f3), bf_cat(b0, b1), acc);
  }
  agg2_epilogue(acc, mbase + aoff, n, B2t, dinv, b2, out);
}

// ---------------------------------------------------------------------------
// Host-side launcher
// ---------------------------------------------------------------------------
static inline size_t align256(size_t x) { return (x + 255) & ~(size_t)255; }

extern "C" void kernel_launch(void* const* d_in, const int* in_sizes, int n_in,
                              void* d_out, int out_size, void* d_ws, size_t ws_size,
                              hipStream_t stream) {
  const float* x   = (const float*)d_in[0];   // [8192,512]
  const float* adj = (const float*)d_in[1];   // [8192,8192]
  const float* W1  = (const float*)d_in[2];   // [512,128]
  const float* b1  = (const float*)d_in[3];   // [128]
  const float* W2  = (const float*)d_in[4];   // [128,16]
  const float* b2  = (const float*)d_in[5];   // [16]
  float* outp      = (float*)d_out;           // [8192,16]
  (void)in_sizes; (void)n_in; (void)out_size;

  // Carve workspace (~12.6 MB base)
  char* w = (char*)d_ws;
  float*  dinv = (float*)w;   w += align256((size_t)NN * 4);
  __bf16* Xh   = (__bf16*)w;  w += align256((size_t)NN * NFEAT * 2);
  __bf16* W1t  = (__bf16*)w;  w += align256((size_t)NHID * NFEAT * 2);
  __bf16* W2t  = (__bf16*)w;  w += align256((size_t)NCLASS * NHID * 2);
  __bf16* B1t  = (__bf16*)w;  w += align256((size_t)NHID * NN * 2);
  __bf16* Hh   = (__bf16*)w;  w += align256((size_t)NN * NHID * 2);
  __bf16* B2t  = (__bf16*)w;  w += align256((size_t)NCLASS * NN * 2);
  size_t base_bytes = (size_t)(w - (char*)d_ws);
  __bf16* Ah   = (__bf16*)w;                       // optional 128 MB bf16 adjacency
  size_t ah_bytes = (size_t)NN * NN * 2;
  bool big = (ws_size >= base_bytes + ah_bytes);   // deterministic path choice

  if (big) {
    k_deg_cvt<<<NN, 256, 0, stream>>>(adj, dinv, Ah);
  } else {
    k_deg    <<<NN, 256, 0, stream>>>(adj, dinv);
  }
  k_cvt_x   <<<(NN * NFEAT) / (8 * 256), 256, 0, stream>>>(x, Xh);
  k_tr_w1   <<<(NFEAT * NHID) / 256,     256, 0, stream>>>(W1, W1t);
  k_tr_w2   <<<(NHID * NCLASS) / 256,    256, 0, stream>>>(W2, W2t);
  k_gemm_xw1<<<NN / 16,                  256, 0, stream>>>(Xh, W1t, dinv, B1t);
  if (big) {
    k_agg1_bf<<<NN / 16,                 256, 0, stream>>>(Ah, B1t, dinv, b1, Hh);
  } else {
    k_agg1   <<<NN / 16,                 256, 0, stream>>>(adj, B1t, dinv, b1, Hh);
  }
  k_gemm_hw2<<<NN / 128,                 256, 0, stream>>>(Hh, W2t, dinv, B2t);
  if (big) {
    k_agg2_bf<<<NN / 128,                256, 0, stream>>>(Ah, B2t, dinv, b2, outp);
  } else {
    k_agg2   <<<NN / 128,                256, 0, stream>>>(adj, B2t, dinv, b2, outp);
  }
}
